// KNOConv2d_283467841870
// MI455X (gfx1250) — matrix operand order repaired and checked by reading, not compiled, not run on hardware
//
#include <hip/hip_runtime.h>
#include <hip/hip_bf16.h>
#include <math.h>

// ---------------------------------------------------------------------------
// KNO spectral conv on MI455X (gfx1250, wave32, WMMA, async-LDS DMA).
//   1) k_dft_w  : Y = x @ F1                (M=65536,K=256,N=64)  f16 WMMA
//   2) k_dft_h  : Z = E2 @ Y                (per bc: 128x512x32)
//   3) k_mix    : per-mode complex channel mix, r iterations
//   4) k_idft_h : G = E4 @ Z2               (per bo: 256x128x64)
//   5) k_idft_w : out = G @ C4 + x          (per bo: 256x64x256)
// DFT basis tables are precomputed PRE-PACKED in WMMA fragment layout so
// each wave loads its matrix fragment with one b128-pair straight from L2.
// ---------------------------------------------------------------------------

typedef __attribute__((ext_vector_type(16))) _Float16 v16h;
typedef __attribute__((ext_vector_type(8)))  float    v8f;
typedef __attribute__((ext_vector_type(4)))  int      i32x4;

#define PI_F 3.14159265358979323846f

// ---- workspace byte offsets -----------------------------------------------
static constexpr size_t OFF_F1 = 0;                       // packed-B [8kt][4nt]
static constexpr size_t OFF_E2 = OFF_F1 + 256*64*2;       // packed-A [8mt][16kt]
static constexpr size_t OFF_E4 = OFF_E2 + 128*512*2;      // packed-A [16mt][4kt]
static constexpr size_t OFF_C4 = OFF_E4 + 256*128*2;      // packed-B [2kt][16nt]
static constexpr size_t OFF_Y  = OFF_C4 + 64*256*2;       // [256][512][32] f16
static constexpr size_t OFF_Z  = OFF_Y  + (size_t)256*512*32*2;  // [256][128][32]
static constexpr size_t OFF_Z2 = OFF_Z  + (size_t)256*128*32*2;  // [256][128][64]
static constexpr size_t OFF_G  = OFF_Z2 + (size_t)256*128*64*2;  // [256][256][64]

// ---- async global->LDS DMA (ASYNCcnt path), with sync fallback ------------
#if __has_builtin(__builtin_amdgcn_global_load_async_to_lds_b128)
#define HAS_ASYNC_LDS 1
#else
#define HAS_ASYNC_LDS 0
#endif

__device__ __forceinline__ void cp16(void* lds, const void* g) {
#if HAS_ASYNC_LDS
  __builtin_amdgcn_global_load_async_to_lds_b128((i32x4*)g, (i32x4*)lds, 0, 0);
#else
  *(uint4*)lds = *(const uint4*)g;
#endif
}
__device__ __forceinline__ void cp_wait() {
#if HAS_ASYNC_LDS
#if __has_builtin(__builtin_amdgcn_s_wait_asynccnt)
  __builtin_amdgcn_s_wait_asynccnt(0);
#else
  asm volatile("s_wait_asynccnt 0" ::: "memory");
#endif
#endif
}

// ---- WMMA fragment helpers (CDNA5 16x16x32 f16 layouts, wave32) -----------
// A (16x32) from LDS: lane<16: row M=lane, K=0..15; lane>=16: K=16..31
__device__ __forceinline__ v16h frag_a(const _Float16* base, int ld) {
  const int l = threadIdx.x & 31;
  const _Float16* row = base + (l & 15) * ld + ((l >> 4) << 4);
  v16h a;
#pragma unroll
  for (int e = 0; e < 16; ++e) a[e] = row[e];
  return a;
}
// B (32x16) from LDS: lane<16: col N=lane, K=0..15; lane>=16: K=16..31
__device__ __forceinline__ v16h frag_b(const _Float16* base, int ld) {
  const int l = threadIdx.x & 31;
  const _Float16* col = base + ((l >> 4) << 4) * ld + (l & 15);
  v16h b;
#pragma unroll
  for (int e = 0; e < 16; ++e) b[e] = col[e * ld];
  return b;
}
// fragment pre-packed in register layout: one 32B load per lane
__device__ __forceinline__ v16h frag_packed(const _Float16* tab, int frag) {
  return *(const v16h*)(tab + (((size_t)frag * 32 + (threadIdx.x & 31)) << 4));
}
__device__ __forceinline__ v8f wmma_f16(v16h a, v16h b, v8f c) {
  return __builtin_amdgcn_wmma_f32_16x16x32_f16(false, a, false, b, (short)0, c,
                                                false, false);
}

// ---- basis generation (packed fragment layouts) ---------------------------
__global__ void k_init_basis(_Float16* F1, _Float16* E2, _Float16* E4,
                             _Float16* C4) {
  const int id = blockIdx.x * blockDim.x + threadIdx.x;
  const int stride = gridDim.x * blockDim.x;
  const float s = 1.0f / 16.0f;  // ortho: 1/sqrt(256) per transform dimension
  // F1 packed-B: K=256(w), N=64; frag f = ktile*4 + nt
  for (int t = id; t < 16384; t += stride) {
    int f = t >> 9, lane = (t >> 4) & 31, e = t & 15;
    int ktile = f >> 2, nt = f & 3;
    int w = ktile * 32 + ((lane >> 4) << 4) + e;
    int n = nt * 16 + (lane & 15), ky = n & 31;
    float th = 2.f * PI_F * (float)(ky * w) * (1.f / 256.f);
    F1[t] = (_Float16)(((n < 32) ? cosf(th) : -sinf(th)) * s);
  }
  // E2 packed-A: M=128, K=512; frag f = mrow*16 + ktile
  // rows 0..63 -> Zr(kx), 64..127 -> Zi ; cols 0..255 Yr(h), 256..511 Yi
  for (int t = id; t < 65536; t += stride) {
    int f = t >> 9, lane = (t >> 4) & 31, e = t & 15;
    int mrow = f >> 4, ktile = f & 15;
    int m = mrow * 16 + (lane & 15);
    int k = ktile * 32 + ((lane >> 4) << 4) + e;
    int kx = m & 63, kxg = (kx < 32) ? kx : (192 + kx);
    int h = k & 255;
    float th = 2.f * PI_F * (float)(kxg * h) * (1.f / 256.f);
    float cs = cosf(th) * s, sn = sinf(th) * s;
    E2[t] = (_Float16)((m < 64) ? ((k < 256) ? cs : sn)
                                : ((k < 256) ? -sn : cs));
  }
  // E4 packed-A: M=256(h), K=128; frag f = mrow*4 + ktile; k<64 cos, else sin
  for (int t = id; t < 32768; t += stride) {
    int f = t >> 9, lane = (t >> 4) & 31, e = t & 15;
    int mrow = f >> 2, ktile = f & 3;
    int m = mrow * 16 + (lane & 15);
    int k = ktile * 32 + ((lane >> 4) << 4) + e;
    int kx = k & 63, kxg = (kx < 32) ? kx : (192 + kx);
    float th = 2.f * PI_F * (float)(kxg * m) * (1.f / 256.f);
    E4[t] = (_Float16)(((k < 64) ? cosf(th) : sinf(th)) * s);
  }
  // C4 packed-B: K=64, N=256(w); frag f = ktile*16 + ntile
  for (int t = id; t < 16384; t += stride) {
    int f = t >> 9, lane = (t >> 4) & 31, e = t & 15;
    int ktile = f >> 4, ntile = f & 15;
    int k = ktile * 32 + ((lane >> 4) << 4) + e;
    int n = ntile * 16 + (lane & 15), ky = k & 31;
    float th = 2.f * PI_F * (float)(ky * n) * (1.f / 256.f);
    float v = (k < 32) ? ((ky == 0) ? 1.f : 2.f) * cosf(th)
                       : ((ky == 0) ? 0.f : -2.f) * sinf(th);
    C4[t] = (_Float16)(v * s);
  }
}

// ---- stage 1: forward DFT along w -----------------------------------------
__global__ __launch_bounds__(128)
void k_dft_w(const float* __restrict__ x, const _Float16* __restrict__ F1,
             _Float16* __restrict__ Y) {
  alignas(16) __shared__ _Float16 As[64 * 48];
  alignas(16) __shared__ _Float16 Ot[64 * 72];
  const int tid = threadIdx.x, wave = tid >> 5;
  const int m0 = blockIdx.x * 64;
  v8f acc[4] = {};
  for (int kk = 0; kk < 256; kk += 32) {
    // A tile: 64x32 f32 -> f16, fully vectorized (b128 in, b128 to LDS)
    {
      int row = tid >> 1, half = tid & 1;
      const float4* s4 =
          (const float4*)(x + (size_t)(m0 + row) * 256 + kk + half * 16);
      union { float f[16]; float4 v[4]; } in;
#pragma unroll
      for (int i = 0; i < 4; ++i) in.v[i] = s4[i];
      union { _Float16 h[16]; uint4 u[2]; } pk;
#pragma unroll
      for (int e = 0; e < 16; ++e) pk.h[e] = (_Float16)in.f[e];
      uint4* dst = (uint4*)&As[row * 48 + half * 16];
      dst[0] = pk.u[0];
      dst[1] = pk.u[1];
    }
    if (kk + 32 < 256)  // stream next A tile toward cache
      __builtin_prefetch(&x[(size_t)(m0 + (tid >> 1)) * 256 + kk + 32 +
                            (tid & 1) * 16], 0, 1);
    __syncthreads();
    v16h a = frag_a(&As[wave * 16 * 48], 48);
    const int kt = kk >> 5;
#pragma unroll
    for (int nt = 0; nt < 4; ++nt)
      acc[nt] = wmma_f16(a, frag_packed(F1, kt * 4 + nt), acc[nt]);
    __syncthreads();
  }
  // epilogue: transpose via LDS, store coalesced b128 rows of Y
  const int lane = tid & 31, hi = lane >> 4, nn = lane & 15;
#pragma unroll
  for (int nt = 0; nt < 4; ++nt)
#pragma unroll
    for (int v = 0; v < 8; ++v)
      Ot[(wave * 16 + v + hi * 8) * 72 + nt * 16 + nn] = (_Float16)acc[nt][v];
  __syncthreads();
  {
    int m = tid & 63, split = tid >> 6;
    int gm = m0 + m, bc = gm >> 8, h = gm & 255;
    uint4* dst = (uint4*)(Y + (size_t)(bc * 512 + split * 256 + h) * 32);
    const uint4* src = (const uint4*)&Ot[m * 72 + split * 32];
#pragma unroll
    for (int i = 0; i < 4; ++i) dst[i] = src[i];
  }
}

// ---- stage 2: forward DFT along h (per (b,c)) -----------------------------
__global__ __launch_bounds__(128)
void k_dft_h(const _Float16* __restrict__ E2, const _Float16* __restrict__ Y,
             _Float16* __restrict__ Z) {
  alignas(16) __shared__ _Float16 Bs[32 * 40];
  alignas(16) __shared__ _Float16 Ot[128 * 40];
  const int tid = threadIdx.x, wave = tid >> 5, bc = blockIdx.x;
  v8f acc[2][2] = {};
  for (int kk = 0; kk < 512; kk += 32) {
    {  // B tile 32x32 via async DMA to LDS (16B chunks)
      int row = tid >> 2, seg = tid & 3;
      cp16(&Bs[row * 40 + seg * 8],
           Y + (size_t)(bc * 512 + kk + row) * 32 + seg * 8);
    }
    cp_wait();
    __syncthreads();
    const int kt = kk >> 5;
#pragma unroll
    for (int mt = 0; mt < 2; ++mt) {
      v16h a = frag_packed(E2, (wave * 2 + mt) * 16 + kt);
#pragma unroll
      for (int nt = 0; nt < 2; ++nt)
        acc[mt][nt] = wmma_f16(a, frag_b(&Bs[nt * 16], 40), acc[mt][nt]);
    }
    __syncthreads();
  }
  const int lane = tid & 31, hi = lane >> 4, nn = lane & 15;
#pragma unroll
  for (int mt = 0; mt < 2; ++mt)
#pragma unroll
    for (int nt = 0; nt < 2; ++nt)
#pragma unroll
      for (int v = 0; v < 8; ++v)
        Ot[((wave * 2 + mt) * 16 + v + hi * 8) * 40 + nt * 16 + nn] =
            (_Float16)acc[mt][nt][v];
  __syncthreads();
  {
    uint4* dst = (uint4*)(Z + (size_t)(bc * 128 + tid) * 32);
    const uint4* src = (const uint4*)&Ot[tid * 40];
#pragma unroll
    for (int i = 0; i < 4; ++i) dst[i] = src[i];
  }
}

// ---- stage 3: per-mode complex channel mixing, applied r times ------------
__global__ __launch_bounds__(32)
void k_mix(const float2* __restrict__ wgt, const _Float16* __restrict__ Z,
           const int* __restrict__ rr, _Float16* __restrict__ Z2) {
  alignas(16) __shared__ _Float16 Wr[32 * 33], Wi[32 * 33], Wn[32 * 33];
  alignas(16) __shared__ _Float16 Za[16 * 64];
  const int lane = threadIdx.x;
  const int kx = blockIdx.x, ky = blockIdx.y;
  const int xw = kx & 31;  // bot block (kx>=32) reuses Wblk rows 0..31
  for (int i = 0; i < 32; ++i) {
    float2 wv = wgt[(size_t)((i * 32 + lane) * 32 + xw) * 32 + ky];
    Wr[i * 33 + lane] = (_Float16)wv.x;
    Wi[i * 33 + lane] = (_Float16)wv.y;
    Wn[i * 33 + lane] = (_Float16)(-wv.y);
  }
  for (int b = 0; b < 8; ++b) {  // A = [zr | zi] K-stacked, rows = batch
    Za[b * 64 + lane]      = Z[(size_t)((b * 32 + lane) * 128 + kx) * 32 + ky];
    Za[b * 64 + 32 + lane] = Z[(size_t)((b * 32 + lane) * 128 + 64 + kx) * 32 + ky];
  }
  for (int rw = 8; rw < 16; ++rw) {  // pad M to 16
    Za[rw * 64 + lane] = (_Float16)0.f;
    Za[rw * 64 + 32 + lane] = (_Float16)0.f;
  }
  __syncthreads();
  const int r = rr[0];
  for (int it = 0; it < r; ++it) {
    v16h alo = frag_a(Za, 64);        // zr (K=0..31)
    v16h ahi = frag_a(Za + 32, 64);   // zi (K=32..63)
    v8f orr[2], oii[2];
#pragma unroll
    for (int nt = 0; nt < 2; ++nt) {
      v8f z0 = {};
      // or = zr@Wr - zi@Wi ; oi = zr@Wi + zi@Wr
      orr[nt] = wmma_f16(ahi, frag_b(&Wn[nt * 16], 33),
                         wmma_f16(alo, frag_b(&Wr[nt * 16], 33), z0));
      oii[nt] = wmma_f16(ahi, frag_b(&Wr[nt * 16], 33),
                         wmma_f16(alo, frag_b(&Wi[nt * 16], 33), z0));
    }
    __syncthreads();
    const int hi = lane >> 4, nn = lane & 15;
#pragma unroll
    for (int nt = 0; nt < 2; ++nt)
#pragma unroll
      for (int v = 0; v < 8; ++v) {
        int m = v + hi * 8;
        if (m < 8) {
          Za[m * 64 + nt * 16 + nn]      = (_Float16)orr[nt][v];
          Za[m * 64 + 32 + nt * 16 + nn] = (_Float16)oii[nt][v];
        }
      }
    __syncthreads();
  }
  // scatter to Z2[bo][128][64] = [[Zr, Zi], [-Zi, Zr]] (B operand of k_idft_h)
  for (int b = 0; b < 8; ++b) {
    _Float16 re = Za[b * 64 + lane];
    _Float16 im = Za[b * 64 + 32 + lane];
    size_t base = (size_t)((b * 32 + lane) * 128) * 64;
    Z2[base + (size_t)kx * 64 + ky]             = re;
    Z2[base + (size_t)kx * 64 + 32 + ky]        = im;
    Z2[base + (size_t)(64 + kx) * 64 + ky]      = (_Float16)(-(float)im);
    Z2[base + (size_t)(64 + kx) * 64 + 32 + ky] = re;
  }
}

// ---- stage 4a: inverse DFT along h (per (b,o)) ----------------------------
__global__ __launch_bounds__(128)
void k_idft_h(const _Float16* __restrict__ E4, const _Float16* __restrict__ Z2,
              _Float16* __restrict__ G) {
  alignas(16) __shared__ _Float16 Bs[32 * 72];
  alignas(16) __shared__ _Float16 Ot[64 * 72];
  const int tid = threadIdx.x, wave = tid >> 5;
  const int bo = blockIdx.y, m0 = blockIdx.x * 64;
  v8f acc[4] = {};
  for (int kk = 0; kk < 128; kk += 32) {
    {  // B tile 32x64 via async DMA (256 x 16B chunks)
#pragma unroll
      for (int i = 0; i < 2; ++i) {
        int c = tid + i * 128, row = c >> 3, seg = c & 7;
        cp16(&Bs[row * 72 + seg * 8],
             Z2 + ((size_t)bo * 128 + kk + row) * 64 + seg * 8);
      }
    }
    cp_wait();
    __syncthreads();
    const int kt = kk >> 5;
    v16h a = frag_packed(E4, (blockIdx.x * 4 + wave) * 4 + kt);
#pragma unroll
    for (int nt = 0; nt < 4; ++nt)
      acc[nt] = wmma_f16(a, frag_b(&Bs[nt * 16], 72), acc[nt]);
    __syncthreads();
  }
  const int lane = tid & 31, hi = lane >> 4, nn = lane & 15;
#pragma unroll
  for (int nt = 0; nt < 4; ++nt)
#pragma unroll
    for (int v = 0; v < 8; ++v)
      Ot[(wave * 16 + v + hi * 8) * 72 + nt * 16 + nn] = (_Float16)acc[nt][v];
  __syncthreads();
  {
    int row = tid >> 1, half = tid & 1;
    uint4* dst = (uint4*)(G + ((size_t)bo * 256 + m0 + row) * 64 + half * 32);
    const uint4* src = (const uint4*)&Ot[row * 72 + half * 32];
#pragma unroll
    for (int i = 0; i < 4; ++i) dst[i] = src[i];
  }
}

// ---- stage 4b: inverse rDFT along w + residual ----------------------------
__global__ __launch_bounds__(128)
void k_idft_w(const _Float16* __restrict__ G, const _Float16* __restrict__ C4,
              const float* __restrict__ x, float* __restrict__ out) {
  alignas(16) __shared__ _Float16 As[64 * 48];
  alignas(16) __shared__ float Of[64 * 68];
  const int tid = threadIdx.x, wave = tid >> 5;
  const int bo = blockIdx.y;
  const int m0 = (blockIdx.x & 3) * 64, n0 = (blockIdx.x >> 2) * 64;
  v8f acc[4] = {};
  for (int kk = 0; kk < 64; kk += 32) {
    {  // A tile 64x32 via async DMA (256 x 16B chunks)
#pragma unroll
      for (int i = 0; i < 2; ++i) {
        int c = tid + i * 128, row = c >> 2, seg = c & 3;
        cp16(&As[row * 48 + seg * 8],
             G + ((size_t)bo * 256 + m0 + row) * 64 + kk + seg * 8);
      }
    }
    cp_wait();
    __syncthreads();
    const int kt = kk >> 5;
    v16h a = frag_a(&As[wave * 16 * 48], 48);
#pragma unroll
    for (int nt = 0; nt < 4; ++nt)
      acc[nt] = wmma_f16(
          a, frag_packed(C4, kt * 16 + (blockIdx.x >> 2) * 4 + nt), acc[nt]);
    __syncthreads();
  }
  const int lane = tid & 31, hi = lane >> 4, nn = lane & 15;
#pragma unroll
  for (int nt = 0; nt < 4; ++nt)
#pragma unroll
    for (int v = 0; v < 8; ++v)
      Of[(wave * 16 + v + hi * 8) * 68 + nt * 16 + nn] = acc[nt][v];
  __syncthreads();
  {  // residual add + coalesced float4 stores
    int row = tid >> 1, half = tid & 1;
    size_t gidx = (size_t)bo * 65536 + (size_t)(m0 + row) * 256 + n0 + half * 32;
    const float4* sv = (const float4*)&Of[row * 68 + half * 32];
    const float4* xv = (const float4*)(x + gidx);
    float4* ov = (float4*)(out + gidx);
#pragma unroll
    for (int i = 0; i < 8; ++i) {
      float4 a4 = sv[i], b4 = xv[i];
      a4.x += b4.x; a4.y += b4.y; a4.z += b4.z; a4.w += b4.w;
      ov[i] = a4;
    }
  }
}

// ---------------------------------------------------------------------------
extern "C" void kernel_launch(void* const* d_in, const int* in_sizes, int n_in,
                              void* d_out, int out_size, void* d_ws,
                              size_t ws_size, hipStream_t stream) {
  (void)in_sizes; (void)n_in; (void)out_size; (void)ws_size;
  const float*  x  = (const float*)d_in[0];
  const float2* wg = (const float2*)d_in[1];   // complex64 weight (re,im)
  const int*    r  = (const int*)d_in[2];
  float* out = (float*)d_out;
  char* ws = (char*)d_ws;

  _Float16* F1 = (_Float16*)(ws + OFF_F1);
  _Float16* E2 = (_Float16*)(ws + OFF_E2);
  _Float16* E4 = (_Float16*)(ws + OFF_E4);
  _Float16* C4 = (_Float16*)(ws + OFF_C4);
  _Float16* Y  = (_Float16*)(ws + OFF_Y);
  _Float16* Z  = (_Float16*)(ws + OFF_Z);
  _Float16* Z2 = (_Float16*)(ws + OFF_Z2);
  _Float16* G  = (_Float16*)(ws + OFF_G);

  k_init_basis<<<dim3(256), dim3(256), 0, stream>>>(F1, E2, E4, C4);
  k_dft_w <<<dim3(1024),    dim3(128), 0, stream>>>(x, F1, Y);
  k_dft_h <<<dim3(256),     dim3(128), 0, stream>>>(E2, Y, Z);
  k_mix   <<<dim3(64, 32),  dim3(32),  0, stream>>>(wg, Z, r, Z2);
  k_idft_h<<<dim3(4, 256),  dim3(128), 0, stream>>>(E4, Z2, G);
  k_idft_w<<<dim3(16, 256), dim3(128), 0, stream>>>(G, C4, x, out);
}
